// GINNet_73280732004448
// MI455X (gfx1250) — matrix operand order repaired and checked by reading, not compiled, not run on hardware
//
#include <hip/hip_runtime.h>
#include <math.h>

typedef __attribute__((ext_vector_type(16))) _Float16 v16h;
typedef __attribute__((ext_vector_type(8)))  float    v8f;
typedef __attribute__((ext_vector_type(4)))  float    v4f;

#define DIM      32
#define NCLS     10
#define BN_EPS   1e-5f

// ---------------------------------------------------------------------------
// Fragment builders matching CDNA5 wave32 WMMA 16x16x32 f16 VGPR layouts.
// A (16x32, MxK): lane m (0-15) row m holds K in {0..7, 16..23};
//                 lane m+16 row m holds K in {8..15, 24..31}.
// a[2v],a[2v+1] = the two f16 halves of VGPR v.
// ---------------------------------------------------------------------------
__device__ __forceinline__ v16h a_frag_from_f32(const float* __restrict__ act,
                                                int row, int ld, int kbase,
                                                bool valid) {
  v16h a;
  const int t    = threadIdx.x & 31;
  const int half = t >> 4;
  if (valid) {
    const float* r = act + (size_t)row * ld + kbase + half * 8;
    v4f x0 = *(const v4f*)(r);        // K rel 0..3
    v4f x1 = *(const v4f*)(r + 4);    // K rel 4..7
    v4f x2 = *(const v4f*)(r + 16);   // K rel 16..19 (abs 16+half*8 ..)
    v4f x3 = *(const v4f*)(r + 20);   // K rel 20..23
#pragma unroll
    for (int i = 0; i < 4; ++i) {
      a[i]      = (_Float16)x0[i];
      a[4 + i]  = (_Float16)x1[i];
      a[8 + i]  = (_Float16)x2[i];
      a[12 + i] = (_Float16)x3[i];
    }
  } else {
#pragma unroll
    for (int i = 0; i < 16; ++i) a[i] = (_Float16)0.0f;
  }
  return a;
}

// B (32x16, KxN) from row-major W[K][ldw], columns c0..c0+15:
// lane n (0-15): col c0+n, K in {0..7, 16..23};
// lane n+16:     col c0+n, K in {8..15, 24..31}.
__device__ __forceinline__ v16h b_frag_from_w(const float* __restrict__ W,
                                              int ldw, int kbase, int c0) {
  v16h b;
  const int t    = threadIdx.x & 31;
  const int half = t >> 4;
  const int n    = t & 15;
  const float* p = W + (size_t)(kbase + half * 8) * ldw + c0 + n;
#pragma unroll
  for (int v = 0; v < 8; ++v) {
    b[v]     = (_Float16)p[v * ldw];
    b[8 + v] = (_Float16)p[(16 + v) * ldw];
  }
  return b;
}

// ---------------------------------------------------------------------------
// out[n x 32] = act[n x K] @ W[K x 32]      (K multiple of 32; f32 accum WMMA)
// One wave -> 16 rows; block 256 -> 128 rows.
// ---------------------------------------------------------------------------
__global__ __launch_bounds__(256)
void gemm_nx32(const float* __restrict__ act, const float* __restrict__ W,
               float* __restrict__ out, int n, int K) {
  const int wave = (int)((blockIdx.x * blockDim.x + threadIdx.x) >> 5);
  const int row0 = wave * 16;
  if (row0 >= n) return;                       // uniform per wave
  const int t    = threadIdx.x & 31;
  const int half = t >> 4;
  const int m    = t & 15;
  v8f c0 = {}, c1 = {};
  for (int kb = 0; kb < K; kb += 32) {
    v16h a  = a_frag_from_f32(act, row0 + m, K, kb, (row0 + m) < n);
    v16h bl = b_frag_from_w(W, DIM, kb, 0);
    v16h bh = b_frag_from_w(W, DIM, kb, 16);
    c0 = __builtin_amdgcn_wmma_f32_16x16x32_f16(false, a, false, bl, (short)0, c0, false, false);
    c1 = __builtin_amdgcn_wmma_f32_16x16x32_f16(false, a, false, bh, (short)0, c1, false, false);
  }
#pragma unroll
  for (int j = 0; j < 8; ++j) {
    const int r = row0 + j + half * 8;
    if (r < n) {
      out[(size_t)r * DIM + m]      = c0[j];
      out[(size_t)r * DIM + 16 + m] = c1[j];
    }
  }
}

// ---------------------------------------------------------------------------
// agg[dst] += q[src], 32 feats/edge, 8 lanes per edge, float4 + 4 atomics.
// ---------------------------------------------------------------------------
__global__ __launch_bounds__(256)
void scatter_add32(const int* __restrict__ ei, const float* __restrict__ q,
                   float* __restrict__ agg, int n_edges) {
  const long long gid = (long long)blockIdx.x * blockDim.x + threadIdx.x;
  if (gid >= (long long)n_edges * 8) return;
  const int e    = (int)(gid >> 3);
  const int part = (int)(gid & 7);
  const int s = ei[e];
  const int d = ei[n_edges + e];
  v4f val = *(const v4f*)(q + (size_t)s * DIM + part * 4);
  float* dp = agg + (size_t)d * DIM + part * 4;
  atomicAdd(dp + 0, val[0]);
  atomicAdd(dp + 1, val[1]);
  atomicAdd(dp + 2, val[2]);
  atomicAdd(dp + 3, val[3]);
}

__global__ __launch_bounds__(256)
void zero_v4(float* __restrict__ p, long long n4) {
  const long long i = (long long)blockIdx.x * blockDim.x + threadIdx.x;
  if (i < n4) {
    v4f z = {0.0f, 0.0f, 0.0f, 0.0f};
    ((v4f*)p)[i] = z;
  }
}

// ---------------------------------------------------------------------------
// Fused GIN MLP tail: h = BN(relu( relu(q + agg + ba) @ Wb + bb ))
// ---------------------------------------------------------------------------
__global__ __launch_bounds__(256)
void gin_mlp(const float* __restrict__ q, const float* __restrict__ agg,
             const float* __restrict__ ba, const float* __restrict__ Wb,
             const float* __restrict__ bb,
             const float* __restrict__ gam, const float* __restrict__ bet,
             const float* __restrict__ rm,  const float* __restrict__ rv,
             float* __restrict__ out, int n) {
  const int wave = (int)((blockIdx.x * blockDim.x + threadIdx.x) >> 5);
  const int row0 = wave * 16;
  if (row0 >= n) return;
  const int t    = threadIdx.x & 31;
  const int half = t >> 4;
  const int m    = t & 15;

  // Build A = relu(q + agg + ba) fragment (row-major, K = 32).
  v16h a;
  const int  row   = row0 + m;
  const bool valid = row < n;
  if (valid) {
    const float* pq = q   + (size_t)row * DIM + half * 8;
    const float* pg = agg + (size_t)row * DIM + half * 8;
    const float* pb = ba  + half * 8;
    v4f q0 = *(const v4f*)(pq);      v4f g0 = *(const v4f*)(pg);      v4f b0 = *(const v4f*)(pb);
    v4f q1 = *(const v4f*)(pq + 4);  v4f g1 = *(const v4f*)(pg + 4);  v4f b1 = *(const v4f*)(pb + 4);
    v4f q2 = *(const v4f*)(pq + 16); v4f g2 = *(const v4f*)(pg + 16); v4f b2 = *(const v4f*)(pb + 16);
    v4f q3 = *(const v4f*)(pq + 20); v4f g3 = *(const v4f*)(pg + 20); v4f b3 = *(const v4f*)(pb + 20);
#pragma unroll
    for (int i = 0; i < 4; ++i) {
      a[i]      = (_Float16)fmaxf(q0[i] + g0[i] + b0[i], 0.0f);
      a[4 + i]  = (_Float16)fmaxf(q1[i] + g1[i] + b1[i], 0.0f);
      a[8 + i]  = (_Float16)fmaxf(q2[i] + g2[i] + b2[i], 0.0f);
      a[12 + i] = (_Float16)fmaxf(q3[i] + g3[i] + b3[i], 0.0f);
    }
  } else {
#pragma unroll
    for (int i = 0; i < 16; ++i) a[i] = (_Float16)0.0f;
  }

  v16h bl = b_frag_from_w(Wb, DIM, 0, 0);
  v16h bh = b_frag_from_w(Wb, DIM, 0, 16);
  v8f c0 = {}, c1 = {};
  c0 = __builtin_amdgcn_wmma_f32_16x16x32_f16(false, a, false, bl, (short)0, c0, false, false);
  c1 = __builtin_amdgcn_wmma_f32_16x16x32_f16(false, a, false, bh, (short)0, c1, false, false);

  // Epilogue: +bias, relu, BN folded to scale/shift. Column is fixed per lane.
  const int cL = m, cH = 16 + m;
  const float sL = gam[cL] * rsqrtf(rv[cL] + BN_EPS);
  const float tL = bet[cL] - rm[cL] * sL;
  const float sH = gam[cH] * rsqrtf(rv[cH] + BN_EPS);
  const float tH = bet[cH] - rm[cH] * sH;
  const float bbL = bb[cL], bbH = bb[cH];
#pragma unroll
  for (int j = 0; j < 8; ++j) {
    const int r = row0 + j + half * 8;
    if (r < n) {
      out[(size_t)r * DIM + cL] = fmaxf(c0[j] + bbL, 0.0f) * sL + tL;
      out[(size_t)r * DIM + cH] = fmaxf(c1[j] + bbH, 0.0f) * sH + tH;
    }
  }
}

// ---------------------------------------------------------------------------
// Head: logits = h @ Wfc + bfc ; out = log_softmax(logits)
// ---------------------------------------------------------------------------
__global__ __launch_bounds__(256)
void head_logsoftmax(const float* __restrict__ h, const float* __restrict__ Wfc,
                     const float* __restrict__ bfc, float* __restrict__ out, int n) {
  const int i = blockIdx.x * blockDim.x + threadIdx.x;
  if (i >= n) return;
  float hv[DIM];
#pragma unroll
  for (int k = 0; k < DIM; ++k) hv[k] = h[(size_t)i * DIM + k];
  float lg[NCLS];
  float mx = -INFINITY;
#pragma unroll
  for (int c = 0; c < NCLS; ++c) {
    float s = bfc[c];
#pragma unroll
    for (int k = 0; k < DIM; ++k) s = fmaf(hv[k], Wfc[k * NCLS + c], s);
    lg[c] = s;
    mx = fmaxf(mx, s);
  }
  float se = 0.0f;
#pragma unroll
  for (int c = 0; c < NCLS; ++c) se += expf(lg[c] - mx);
  const float lse = logf(se);
#pragma unroll
  for (int c = 0; c < NCLS; ++c) out[(size_t)i * NCLS + c] = lg[c] - mx - lse;
}

// ---------------------------------------------------------------------------
extern "C" void kernel_launch(void* const* d_in, const int* in_sizes, int n_in,
                              void* d_out, int out_size, void* d_ws, size_t ws_size,
                              hipStream_t stream) {
  const float* x   = (const float*)d_in[0];
  const int*   ei  = (const int*)d_in[1];
  // d_in[2] = batch (unused by the reference output)
  const float* W1a = (const float*)d_in[3];
  const float* b1a = (const float*)d_in[4];
  const float* W1b = (const float*)d_in[5];
  const float* b1b = (const float*)d_in[6];
  const float* W2a = (const float*)d_in[7];
  const float* b2a = (const float*)d_in[8];
  const float* W2b = (const float*)d_in[9];
  const float* b2b = (const float*)d_in[10];
  const float* W3a = (const float*)d_in[11];
  const float* b3a = (const float*)d_in[12];
  const float* W3b = (const float*)d_in[13];
  const float* b3b = (const float*)d_in[14];
  const float* bn1[4] = {(const float*)d_in[15], (const float*)d_in[16],
                         (const float*)d_in[17], (const float*)d_in[18]};
  const float* bn2[4] = {(const float*)d_in[19], (const float*)d_in[20],
                         (const float*)d_in[21], (const float*)d_in[22]};
  const float* bn3[4] = {(const float*)d_in[23], (const float*)d_in[24],
                         (const float*)d_in[25], (const float*)d_in[26]};
  const float* Wfc = (const float*)d_in[27];
  const float* bfc = (const float*)d_in[28];

  const int num_features = 128;
  const int n       = in_sizes[0] / num_features;   // 100000 nodes
  const int n_edges = in_sizes[1] / 2;              // 3200000 edges

  float* q   = (float*)d_ws;                        // n x 32
  float* agg = q + (size_t)n * DIM;                 // n x 32
  float* h   = agg + (size_t)n * DIM;               // n x 32

  const int TPB        = 256;
  const int gemm_grid  = (n + 127) / 128;                         // 128 rows/block
  const long long nv4  = ((long long)n * DIM) / 4;
  const int zero_grid  = (int)((nv4 + TPB - 1) / TPB);
  const int scat_grid  = (int)(((long long)n_edges * 8 + TPB - 1) / TPB);
  const int node_grid  = (n + TPB - 1) / TPB;

  const float* Wa[3] = {W1a, W2a, W3a};
  const float* ba[3] = {b1a, b2a, b3a};
  const float* Wb[3] = {W1b, W2b, W3b};
  const float* bb[3] = {b1b, b2b, b3b};
  const float* const* bn[3] = {bn1, bn2, bn3};

  const float* cur = x;       // current node features
  int curK = num_features;

  for (int L = 0; L < 3; ++L) {
    // q = cur @ Wa  (project BEFORE aggregating: segment_sum commutes with the linear map)
    gemm_nx32<<<gemm_grid, TPB, 0, stream>>>(cur, Wa[L], q, n, curK);
    // agg = segment_sum(q[src] -> dst)
    zero_v4<<<zero_grid, TPB, 0, stream>>>(agg, nv4);
    scatter_add32<<<scat_grid, TPB, 0, stream>>>(ei, q, agg, n_edges);
    // h = BN(relu(relu(q + agg + ba) @ Wb + bb))
    gin_mlp<<<gemm_grid, TPB, 0, stream>>>(q, agg, ba[L], Wb[L], bb[L],
                                           bn[L][0], bn[L][1], bn[L][2], bn[L][3],
                                           h, n);
    cur = h;
    curK = DIM;
  }

  head_logsoftmax<<<node_grid, TPB, 0, stream>>>(h, Wfc, bfc, (float*)d_out, n);
}
